// GLFEncoder_506806141043
// MI455X (gfx1250) — matrix operand order repaired and checked by reading, not compiled, not run on hardware
//
#include <hip/hip_runtime.h>

typedef __attribute__((ext_vector_type(16))) _Float16 v16h;
typedef __attribute__((ext_vector_type(8)))  _Float16 v8h;
typedef __attribute__((ext_vector_type(8)))  float    v8f;
typedef __attribute__((ext_vector_type(4)))  float    v4f;

#define RES    512
#define RANK   8
#define OUT_CH 8
#define ROWH   40   // LDS row stride in halves (80 B: 16B-aligned rows)

__device__ __forceinline__ void sample_plane(const v4f* __restrict__ pl,
                                             float u, float v,
                                             v4f& lo, v4f& hi) {
    // ix = ((u+1)*512 - 1) * 0.5 = u*256 + 255.5   (u already clamped to [-1,1])
    float ix = fmaf(u, 256.0f, 255.5f);
    float iy = fmaf(v, 256.0f, 255.5f);
    float x0f = floorf(ix);
    float y0f = floorf(iy);
    float wx = ix - x0f;
    float wy = iy - y0f;
    int xi = (int)x0f;
    int yi = (int)y0f;
    int x0 = xi     < 0 ? 0 : (xi     > RES - 1 ? RES - 1 : xi);
    int x1 = xi + 1 < 0 ? 0 : (xi + 1 > RES - 1 ? RES - 1 : xi + 1);
    int y0 = yi     < 0 ? 0 : (yi     > RES - 1 ? RES - 1 : yi);
    int y1 = yi + 1 < 0 ? 0 : (yi + 1 > RES - 1 ? RES - 1 : yi + 1);
    // plane viewed as [512][512][2] float4 (8 channels per texel)
    const v4f* r00 = pl + (((y0 << 9) + x0) << 1);
    const v4f* r01 = pl + (((y0 << 9) + x1) << 1);
    const v4f* r10 = pl + (((y1 << 9) + x0) << 1);
    const v4f* r11 = pl + (((y1 << 9) + x1) << 1);
    v4f a0 = r00[0], a1 = r00[1];
    v4f b0 = r01[0], b1 = r01[1];
    v4f c0 = r10[0], c1 = r10[1];
    v4f d0 = r11[0], d1 = r11[1];
    float w00 = (1.0f - wx) * (1.0f - wy);
    float w01 = wx * (1.0f - wy);
    float w10 = (1.0f - wx) * wy;
    float w11 = wx * wy;
    lo = a0 * w00 + b0 * w01 + c0 * w10 + d0 * w11;
    hi = a1 * w00 + b1 * w01 + c1 * w10 + d1 * w11;
}

__device__ __forceinline__ void pack8(v4f lo, v4f hi, _Float16* dst) {
    v8h h;
    h[0] = (_Float16)lo.x; h[1] = (_Float16)lo.y;
    h[2] = (_Float16)lo.z; h[3] = (_Float16)lo.w;
    h[4] = (_Float16)hi.x; h[5] = (_Float16)hi.y;
    h[6] = (_Float16)hi.z; h[7] = (_Float16)hi.w;
    *(v8h*)dst = h;     // 16B-aligned LDS store
}

// Build B (32x16 f16, K=rank, N=point) for wmma half `k` (points k*16..k*16+15).
// Layout: lanes 0-15 hold K=0..15 of point (lane&15); lanes 16-31 hold K=16..31.
__device__ __forceinline__ v16h build_b(const _Float16* lds, int waveBase,
                                        int k, int lane) {
    int row = waveBase + k * 16 + (lane & 15);
    const _Float16* rp = lds + row * ROWH;
    v16h b;
    if (lane < 16) {
        b = *(const v16h*)rp;                 // ranks 0..15 (two ds_load_b128)
    } else {
        v8h t = *(const v8h*)(rp + 16);       // ranks 16..23
        for (int i = 0; i < 8; ++i) b[i] = t[i];
        for (int i = 8; i < 16; ++i) b[i] = (_Float16)0.0f;  // K=24..31 pad
    }
    return b;
}

__global__ __launch_bounds__(256) void glf_triplane_kernel(
    const float* __restrict__ coords,
    const float* __restrict__ plane_xy,
    const float* __restrict__ plane_xz,
    const float* __restrict__ plane_yz,
    const float* __restrict__ proj_w,
    const float* __restrict__ proj_b,
    float* __restrict__ out,
    int N)
{
    __shared__ _Float16 lds[256 * ROWH];   // 20 KB: per-thread f16 feature rows

    const int tid      = threadIdx.x;
    const int lane     = tid & 31;
    const int waveBase = tid & ~31;
    const int p        = blockIdx.x * 256 + tid;
    const int pc       = p < N ? p : N - 1;      // clamp, never early-return

    // ---- A matrix (16x32 f16): M=out channel, K=rank (24 used) -----------
    // lanes 0-15 : M=lane,    halves[0..15]=K0..15
    // lanes 16-31: M=lane-16, halves[0..7]=K16..23, halves[8..15]=K24..31(=0)
    v16h amat;
    {
        for (int i = 0; i < 16; ++i) amat[i] = (_Float16)0.0f;
        int m = lane & 15;
        if (m < OUT_CH) {
            const float* wr = proj_w + m * (3 * RANK);
            if (lane < 16) {
                for (int i = 0; i < 16; ++i) amat[i]    = (_Float16)wr[i];
            } else {
                for (int i = 0; i < 8; ++i) amat[i]     = (_Float16)wr[16 + i];
            }
        }
    }
    // C matrix = bias: lanes 0-15 VGPR r -> channel r; lanes 16-31 unused
    v8f cmat;
    for (int r = 0; r < 8; ++r) cmat[r] = (lane < 16) ? proj_b[r] : 0.0f;

    // ---- gather: clamp coords, bilinear-sample three planes --------------
    float cx = __builtin_nontemporal_load(coords + 3 * pc + 0);
    float cy = __builtin_nontemporal_load(coords + 3 * pc + 1);
    float cz = __builtin_nontemporal_load(coords + 3 * pc + 2);
    cx = fminf(fmaxf(cx, -1.0f), 1.0f);
    cy = fminf(fmaxf(cy, -1.0f), 1.0f);
    cz = fminf(fmaxf(cz, -1.0f), 1.0f);

    v4f f0l, f0h, f1l, f1h, f2l, f2h;
    sample_plane((const v4f*)plane_xy, cx, cy, f0l, f0h);
    sample_plane((const v4f*)plane_xz, cx, cz, f1l, f1h);
    sample_plane((const v4f*)plane_yz, cy, cz, f2l, f2h);

    _Float16* myrow = lds + tid * ROWH;
    pack8(f0l, f0h, myrow + 0);
    pack8(f1l, f1h, myrow + 8);
    pack8(f2l, f2h, myrow + 16);

    // same-wave LDS RAW across lanes: DS pipe is in-order per wave; make the
    // compiler keep order and wait for the stores to land.
    __builtin_amdgcn_wave_barrier();
    asm volatile("s_wait_dscnt 0" ::: "memory");
    __builtin_amdgcn_wave_barrier();

    // ---- projection via WMMA: D[ch][pt] = W x feats^T + b ----------------
    v16h b0 = build_b(lds, waveBase, 0, lane);
    v16h b1 = build_b(lds, waveBase, 1, lane);

    v8f d0 = __builtin_amdgcn_wmma_f32_16x16x32_f16(
        false, amat, false, b0, (short)0, cmat, false, false);
    v8f d1 = __builtin_amdgcn_wmma_f32_16x16x32_f16(
        false, amat, false, b1, (short)0, cmat, false, false);

    // ---- store: lane n<16 owns full 8-ch output of points base+n, base+16+n
    if (lane < 16) {
        int p0 = blockIdx.x * 256 + waveBase + lane;
        int p1 = p0 + 16;
        if (p0 < N) {
            v4f lo, hi;
            lo.x = fminf(fmaxf(d0[0], -10.0f), 10.0f);
            lo.y = fminf(fmaxf(d0[1], -10.0f), 10.0f);
            lo.z = fminf(fmaxf(d0[2], -10.0f), 10.0f);
            lo.w = fminf(fmaxf(d0[3], -10.0f), 10.0f);
            hi.x = fminf(fmaxf(d0[4], -10.0f), 10.0f);
            hi.y = fminf(fmaxf(d0[5], -10.0f), 10.0f);
            hi.z = fminf(fmaxf(d0[6], -10.0f), 10.0f);
            hi.w = fminf(fmaxf(d0[7], -10.0f), 10.0f);
            v4f* dst = (v4f*)(out + (size_t)p0 * 8);
            __builtin_nontemporal_store(lo, dst + 0);
            __builtin_nontemporal_store(hi, dst + 1);
        }
        if (p1 < N) {
            v4f lo, hi;
            lo.x = fminf(fmaxf(d1[0], -10.0f), 10.0f);
            lo.y = fminf(fmaxf(d1[1], -10.0f), 10.0f);
            lo.z = fminf(fmaxf(d1[2], -10.0f), 10.0f);
            lo.w = fminf(fmaxf(d1[3], -10.0f), 10.0f);
            hi.x = fminf(fmaxf(d1[4], -10.0f), 10.0f);
            hi.y = fminf(fmaxf(d1[5], -10.0f), 10.0f);
            hi.z = fminf(fmaxf(d1[6], -10.0f), 10.0f);
            hi.w = fminf(fmaxf(d1[7], -10.0f), 10.0f);
            v4f* dst = (v4f*)(out + (size_t)p1 * 8);
            __builtin_nontemporal_store(lo, dst + 0);
            __builtin_nontemporal_store(hi, dst + 1);
        }
    }
}

extern "C" void kernel_launch(void* const* d_in, const int* in_sizes, int n_in,
                              void* d_out, int out_size, void* d_ws, size_t ws_size,
                              hipStream_t stream) {
    const float* coords = (const float*)d_in[0];
    const float* pxy    = (const float*)d_in[1];
    const float* pxz    = (const float*)d_in[2];
    const float* pyz    = (const float*)d_in[3];
    const float* pw     = (const float*)d_in[4];
    const float* pb     = (const float*)d_in[5];
    float* out          = (float*)d_out;

    int N = in_sizes[0] / 3;
    if (N <= 0) return;
    int blocks = (N + 255) / 256;
    hipLaunchKernelGGL(glf_triplane_kernel, dim3(blocks), dim3(256), 0, stream,
                       coords, pxy, pxz, pyz, pw, pb, out, N);
}
